// BasicBlock_391
// MI455X (gfx1250) — matrix operand order repaired and checked by reading, not compiled, not run on hardware
//
#include <hip/hip_runtime.h>

// ---------------------------------------------------------------------------
// CDNA5 / gfx1250 implementation of a BFP-quantized ResNet BasicBlock.
//   N=64, C=128, H=W=56.  BFP: mant=8 bits, block=64 (flat index).
//   conv runs as implicit GEMM on v_wmma_f32_16x16x32_bf16 (wave32).
// Quantized BFP values (m * 2^(e-7), |m|<=127) are EXACTLY representable in
// bf16, so bf16 WMMA with f32 accumulation matches the fp32 reference conv.
//
// v3b: fix const-ness of halo store pointer. Otherwise identical to v3:
//      two-phase (load-all / store-all) global->LDS staging so loads batch in
//      clauses instead of serializing on one register; all 5 fragments of a
//      tap are loaded before the 4 WMMAs issue, hiding DS latency under XDL.
// ---------------------------------------------------------------------------

typedef __attribute__((ext_vector_type(16))) __bf16        v16bf;
typedef __attribute__((ext_vector_type(8)))  float         v8f;
typedef __attribute__((ext_vector_type(4)))  unsigned int  v4u;

union Frag16 { v4u u[2]; v16bf h; };   // 32B: two 16B LDS loads -> one WMMA operand

#define CC   128      // channels
#define HWD  56       // H == W
#define HW   3136     // 56*56  (= 49 * 64)
#define MT   64       // spatial tile (== BFP block size -> fused requant)
#define KTAP 9

// ---------------------------------------------------------------------------
// BFP quantize: one wave (32 lanes) per 64-element block. out = bf16 (exact).
// ---------------------------------------------------------------------------
__global__ __launch_bounds__(256) void bfp_quant_kernel(
    const float* __restrict__ in, unsigned short* __restrict__ out, int nblocks)
{
    __bf16* ob = (__bf16*)out;
    int lane = threadIdx.x & 31;
    int wgl  = (blockIdx.x << 3) + (threadIdx.x >> 5);
    int nw   = gridDim.x << 3;
    for (int b = wgl; b < nblocks; b += nw) {
        size_t base = (size_t)b * 64;
        float a0 = in[base + lane];
        float a1 = in[base + lane + 32];
        float mx = fmaxf(fabsf(a0), fabsf(a1));
        #pragma unroll
        for (int d = 16; d; d >>= 1) mx = fmaxf(mx, __shfl_xor(mx, d, 32));
        float e   = floorf(log2f(mx + 1e-23f));
        float qs  = exp2f(e - 7.0f);
        float inv = 1.0f / qs;
        float q0 = fminf(fmaxf(rintf(a0 * inv), -127.f), 127.f) * qs;
        float q1 = fminf(fmaxf(rintf(a1 * inv), -127.f), 127.f) * qs;
        ob[base + lane]      = (__bf16)q0;
        ob[base + lane + 32] = (__bf16)q1;
    }
}

// ---------------------------------------------------------------------------
// Weight layout transform: [co][ci][kh][kw] -> [tap][co][ci]  (ci contiguous)
// ---------------------------------------------------------------------------
__global__ __launch_bounds__(256) void wt_transform_kernel(
    const unsigned short* __restrict__ wq, unsigned short* __restrict__ wt)
{
    int i = blockIdx.x * 256 + threadIdx.x;          // over 128*128*9
    if (i >= CC * CC * KTAP) return;
    int tap = i % KTAP;
    int ci  = (i / KTAP) % CC;
    int co  = i / (KTAP * CC);
    wt[((size_t)tap * CC + co) * CC + ci] = wq[i];
}

// ---------------------------------------------------------------------------
// Implicit-GEMM 3x3 conv, pad=1, stride=1, on bf16 WMMA.
//   grid = N*(HW/64) = 3136 WGs x 256 threads (8 wave32).
//   WG tile: M=64 flat spatial x N=128 couts; wave w owns couts [16w,16w+16)
//   with 4 M-subtile accumulators. K loop: 4 ci-chunks x (3 kh x 3 kw) taps.
//   Halo LDS layout [pixel][channel] (stride 40 halfs = 80B: 16B-aligned,
//   conflict-free) => A fragment = 2x ds_load_b128 straight from halo.
// MODE 1: epilogue = BN -> ReLU -> fused BFP requant -> bf16 out
// MODE 2: epilogue = BN -> +residual -> ReLU -> f32 out
// ---------------------------------------------------------------------------
template <int MODE>
__global__ __launch_bounds__(256) void conv3x3_wmma_kernel(
    const unsigned short* __restrict__ xq_u,   // [64][128][3136] bf16
    const unsigned short* __restrict__ wt_u,   // [9][128][128]   bf16
    const float* __restrict__ gamma, const float* __restrict__ beta,
    const float* __restrict__ mean,  const float* __restrict__ var,
    const float* __restrict__ residual,        // MODE 2 only
    void* __restrict__ out)
{
    const __bf16* xq = (const __bf16*)xq_u;
    const __bf16* wt = (const __bf16*)wt_u;

    // LDS: Ahalo [4*58 pixels][40ch] = 18560B ; Bs3 [3*128 rows][40k] = 30720B
    // epilogue aliases the same memory as Qs[128][66] f32 = 33792B
    __shared__ __align__(16) char smem[49280];
    __bf16 (*Ahalo)[40] = (__bf16(*)[40])(smem);            // [r*58+w][c]
    __bf16 (*Bs3)[40]   = (__bf16(*)[40])(smem + 18560);    // [kw*128+co][k]
    float  (*Qs)[66]    = (float(*)[66])(smem);

    const int tid  = threadIdx.x;
    const int wave = tid >> 5;
    const int lane = tid & 31;
    const int l15  = lane & 15;
    const int n      = blockIdx.x / (HW / MT);
    const int mbase  = (blockIdx.x % (HW / MT)) * MT;
    const int hstart = mbase / HWD - 1;          // 4 halo rows cover tile+pad
    const int co0  = wave << 4;
    const int hi   = (lane < 16) ? 0 : 8;        // A-frag K-half select
    const int klo  = (lane < 16) ? 0 : 8;        // B-frag K-half select

    // per-lane A base (biased to dh=dw=-1 so tap offsets are non-negative
    // immediates): halfs offset into Ahalo
    int abase[4];
    #pragma unroll
    for (int t = 0; t < 4; ++t) {
        int sp = mbase + (t << 4) + l15;
        int h0 = sp / HWD, w0 = sp % HWD;
        abase[t] = ((h0 - hstart - 1) * 58 + w0) * 40 + hi;
    }
    __bf16*       ah0w = &Ahalo[0][0];           // write alias (halo staging)
    const __bf16* ah0  = &Ahalo[0][0];           // read alias (fragments)

    v8f acc[4] = {};

    for (int ck = 0; ck < 4; ++ck) {
        __syncthreads();   // previous chunk's halo/B readers are done
        // ---- prefetch next chunk's halo rows (one 128B line per row/channel)
        if (ck < 3 && tid < 128) {
            int cl = tid & 31, r = tid >> 5;
            int h = hstart + r;
            if ((unsigned)h < (unsigned)HWD)
                __builtin_prefetch(
                    xq + ((size_t)n * CC + (ck + 1) * 32 + cl) * HW + h * HWD, 0, 3);
        }
        // ---- stage channel-contiguous halo: 7424 = 29*256 elems.
        //      Two-phase per group of 8: batched coalesced global loads into
        //      distinct regs, then independent scatter ds_store_b16.
        #pragma unroll
        for (int g = 0; g < 4; ++g) {
            __bf16 tmp[8];
            int    loff[8];                       // LDS half-index
            #pragma unroll
            for (int u = 0; u < 8; ++u) {
                int it = g * 8 + u;
                if (it >= 29) break;              // folds at compile time
                int i   = it * 256 + tid;
                int cl  = i / 232;                // channel-in-chunk 0..31
                int rem = i - cl * 232;
                int r   = rem / 58;
                int w   = rem - r * 58 - 1;
                int h   = hstart + r;
                __bf16 v = (__bf16)0.0f;
                if ((unsigned)h < (unsigned)HWD && (unsigned)w < (unsigned)HWD)
                    v = xq[((size_t)n * CC + ck * 32 + cl) * HW + h * HWD + w];
                tmp[u]  = v;
                loff[u] = (r * 58 + (w + 1)) * 40 + cl;
            }
            #pragma unroll
            for (int u = 0; u < 8; ++u) {
                int it = g * 8 + u;
                if (it >= 29) break;
                ah0w[loff[u]] = tmp[u];           // writes into LDS (Ahalo)
            }
        }
        #pragma unroll
        for (int kh = 0; kh < 3; ++kh) {
            __syncthreads();   // prior Bs3 readers done
            // ---- stage 3 taps of B: 1536 16B chunks, 6 per thread, two-phase
            {
                v4u btmp[6];
                #pragma unroll
                for (int it = 0; it < 6; ++it) {
                    int q   = it * 256 + tid;
                    int k16 = (q & 3) << 3;       // 0,8,16,24 halfs
                    int row = q >> 2;             // kw*128 + co
                    int co  = row & 127;
                    int kw  = row >> 7;
                    const __bf16* wp =
                        wt + ((size_t)(kh * 3 + kw) * CC + co) * CC + ck * 32 + k16;
                    btmp[it] = *(const v4u*)wp;
                }
                #pragma unroll
                for (int it = 0; it < 6; ++it) {
                    int q   = it * 256 + tid;
                    int k16 = (q & 3) << 3;
                    int row = q >> 2;
                    *(v4u*)&Bs3[row][k16] = btmp[it];
                }
            }
            __syncthreads();   // halo + Bs3 visible
            #pragma unroll
            for (int kw = 0; kw < 3; ++kw) {
                // load ALL fragments of this tap, then 4 back-to-back WMMAs
                Frag16 fb, fa[4];
                const __bf16* bp = &Bs3[(kw << 7) + co0 + l15][klo];
                fb.u[0] = *(const v4u*)(bp);
                fb.u[1] = *(const v4u*)(bp + 16);
                #pragma unroll
                for (int t = 0; t < 4; ++t) {
                    const __bf16* ap = ah0 + abase[t] + (kh * 58 + kw) * 40;
                    fa[t].u[0] = *(const v4u*)(ap);        // K = hi..hi+7
                    fa[t].u[1] = *(const v4u*)(ap + 16);   // K = 16+hi..23+hi
                }
                #pragma unroll
                for (int t = 0; t < 4; ++t)
                    acc[t] = __builtin_amdgcn_wmma_f32_16x16x32_bf16(
                        false, fa[t].h, false, fb.h, (short)0, acc[t], false, false);
            }
        }
    }

    // ---- epilogue: BN (+ReLU for MODE1) into LDS transpose buffer -----------
    __syncthreads();
    {
        int cog = co0 + l15;
        float sc = gamma[cog] * rsqrtf(var[cog] + 1e-5f);
        float sh = beta[cog] - mean[cog] * sc;
        int hi8 = (lane < 16) ? 0 : 8;
        #pragma unroll
        for (int t = 0; t < 4; ++t)
            #pragma unroll
            for (int v = 0; v < 8; ++v) {
                float val = acc[t][v] * sc + sh;
                if (MODE == 1) val = fmaxf(val, 0.0f);
                Qs[cog][(t << 4) + hi8 + v] = val;
            }
    }
    __syncthreads();

    // ---- coalesced copy-out: thread pair (t, t^1) shares one cout column ----
    int co = tid >> 1;
    int m0 = (tid & 1) << 5;
    size_t base = ((size_t)n * CC + co) * HW + mbase + m0;
    if (MODE == 1) {
        // fused BFP requant: this 64-value column IS one flat BFP block
        float mx = 0.0f;
        #pragma unroll 8
        for (int j = 0; j < 32; ++j) mx = fmaxf(mx, fabsf(Qs[co][m0 + j]));
        mx = fmaxf(mx, __shfl_xor(mx, 1, 32));      // combine the two halves
        float e   = floorf(log2f(mx + 1e-23f));
        float qs  = exp2f(e - 7.0f);
        float inv = 1.0f / qs;
        __bf16* op = (__bf16*)out + base;
        #pragma unroll 8
        for (int j = 0; j < 32; ++j) {
            float q = rintf(Qs[co][m0 + j] * inv);
            op[j] = (__bf16)(fminf(fmaxf(q, -127.f), 127.f) * qs);
        }
    } else {
        const float* rp = residual + base;
        float* op = (float*)out + base;
        #pragma unroll 8
        for (int j = 0; j < 32; ++j)
            op[j] = fmaxf(Qs[co][m0 + j] + rp[j], 0.0f);
    }
}

// ---------------------------------------------------------------------------
extern "C" void kernel_launch(void* const* d_in, const int* in_sizes, int n_in,
                              void* d_out, int out_size, void* d_ws, size_t ws_size,
                              hipStream_t stream)
{
    (void)in_sizes; (void)n_in; (void)out_size; (void)ws_size;
    const float* x  = (const float*)d_in[0];
    const float* w1 = (const float*)d_in[1];
    const float* w2 = (const float*)d_in[2];
    const float* g1 = (const float*)d_in[3];
    const float* b1 = (const float*)d_in[4];
    const float* m1 = (const float*)d_in[5];
    const float* v1 = (const float*)d_in[6];
    const float* g2 = (const float*)d_in[7];
    const float* b2 = (const float*)d_in[8];
    const float* m2 = (const float*)d_in[9];
    const float* v2 = (const float*)d_in[10];

    // workspace carve (all offsets 256B aligned)
    char* ws = (char*)d_ws;
    unsigned short* xq  = (unsigned short*)(ws);                   // 51,380,224 B
    unsigned short* y1q = (unsigned short*)(ws + 51380224);        // 51,380,224 B
    unsigned short* wqt = (unsigned short*)(ws + 102760448);       //    294,912 B
    unsigned short* wt1 = (unsigned short*)(ws + 103055360);       //    294,912 B
    unsigned short* wt2 = (unsigned short*)(ws + 103350272);       //    294,912 B

    const int NXBLK = 64 * CC * HW / 64;   // 401408 BFP blocks in x
    const int NWBLK = CC * CC * KTAP / 64; //   2304 BFP blocks in each weight

    bfp_quant_kernel<<<1024, 256, 0, stream>>>(x, xq, NXBLK);
    bfp_quant_kernel<<<288, 256, 0, stream>>>(w1, wqt, NWBLK);
    wt_transform_kernel<<<576, 256, 0, stream>>>(wqt, wt1);
    bfp_quant_kernel<<<288, 256, 0, stream>>>(w2, wqt, NWBLK);
    wt_transform_kernel<<<576, 256, 0, stream>>>(wqt, wt2);

    // conv1: BN1 -> ReLU -> BFP requant (bf16), fused epilogue
    conv3x3_wmma_kernel<1><<<64 * (HW / MT), 256, 0, stream>>>(
        xq, wt1, g1, b1, m1, v1, nullptr, (void*)y1q);
    // conv2: BN2 -> +residual(x) -> ReLU -> f32 out
    conv3x3_wmma_kernel<2><<<64 * (HW / MT), 256, 0, stream>>>(
        y1q, wt2, g2, b2, m2, v2, x, d_out);
}
